// GaussianSplattingDecoder_14568529068110
// MI455X (gfx1250) — matrix operand order repaired and checked by reading, not compiled, not run on hardware
//
#include <hip/hip_runtime.h>
#include <hip/hip_bf16.h>
#include <math.h>

// ---------------------------------------------------------------------------
// GaussianSplattingDecoder for MI455X (gfx1250, wave32, WMMA)
//
// occ_unnorm[v,c] = sum_k w[v,k] * sem[k,c]   (GEMM M=640000, K=2048, N=18)
// via v_wmma_f32_16x16x32_f16; an all-ones B column produces the normalizer
// sum(w).  Per-pair score math stays f32 VALU (co-executes with XDL WMMA).
// exp(-0.5*mahal)*opacity folded to a single raw v_exp_f32:
//   w = exp2( d2 . (-0.5*log2e*inv_s)  +  log2(opacity) )
// (raw v_exp_f32 is safe: w feeds an f16 convert, which flushes < 2^-25).
// Tables staged to LDS with GLOBAL_LOAD_ASYNC_TO_LDS_B128 (ASYNCcnt path).
// ---------------------------------------------------------------------------

typedef __attribute__((ext_vector_type(16))) _Float16 v16h;
typedef __attribute__((ext_vector_type(8)))  float    v8f;

#define C_CH    17
#define NGAUSS  2048
#define NCHUNK  64          /* 2048 / 32 */
#define NVOX    640000
#define WG_VOX  256         /* 8 waves * 32 voxels */

#if __has_builtin(__builtin_amdgcn_global_load_async_to_lds_b128)
#define USE_ASYNC_LDS 1
// Exact parameter types per hipcc diagnostics:
//   arg0: __attribute__((address_space(1))) int __attribute__((vector_size(16)))*
//   arg1: __attribute__((address_space(3))) int __attribute__((vector_size(16)))*
typedef int async_i4 __attribute__((vector_size(16)));
#define ASYNC_GPTR(p) ((__attribute__((address_space(1))) async_i4*)(p))
#define ASYNC_LPTR(p) ((__attribute__((address_space(3))) async_i4*)(p))
#endif

// Raw v_exp_f32 (no denormal-range fixup; fine since result feeds f16 cvt).
__device__ __forceinline__ float fast_exp2(float x) {
#if __has_builtin(__builtin_amdgcn_exp2f)
    return __builtin_amdgcn_exp2f(x);
#else
    return exp2f(x);
#endif
}

// LDS layout (dynamic):
#define GEOM_OFF   0            /* 2048 * 32B  = 64 KB  */
#define BPRE_OFF   65536        /* 4096 * 32B  = 128 KB */
#define SCR_OFF    196608       /* 8 waves * 32 * 18 f32 = 18432 B */
#define WTS_OFF    215040       /* 1360 f32 padded weights = 5440 B */
#define SMEM_BYTES (215040 + 1360 * 4)

__device__ __forceinline__ float softplus_f(float x) {
    return (x > 20.0f) ? x : log1pf(__expf(x));
}

// ---- Kernel 1: per-gaussian table: [mx,my,mz,ix' | iy',iz',log2(op),0] ----
__global__ void prep_geom(const float* __restrict__ props, float* __restrict__ geomG) {
    int n = blockIdx.x * blockDim.x + threadIdx.x;
    if (n >= NGAUSS) return;
    const float* p = props + n * 28;
    const float NHL2E = -0.72134752044448170f;   // -0.5 * log2(e)
    float sx = softplus_f(p[3]), sy = softplus_f(p[4]), sz = softplus_f(p[5]);
    float ivx = NHL2E / fmaxf(sx * sx, 1e-6f);
    float ivy = NHL2E / fmaxf(sy * sy, 1e-6f);
    float ivz = NHL2E / fmaxf(sz * sz, 1e-6f);
    float lop = -log2f(1.0f + __expf(-p[10]));   // log2(sigmoid(x))
    float4* g = (float4*)(geomG + n * 8);
    g[0] = make_float4(p[0], p[1], p[2], ivx);
    g[1] = make_float4(ivy, ivz, lop, 0.0f);
}

// ---- Kernel 2: pre-swizzled WMMA B operands (f16) per (chunk, n-tile, lane) ----
// B layout (16-bit, 32x16): lanes 0-15 col=lane, K=0..15 (2/VGPR); lanes 16-31
// col=lane-16, K=16..31.  Tile0 cols = channels 0..15; tile1 col0=ch16,
// col1 = 1.0 (weight-sum column), rest 0.
__global__ void prep_bmat(const float* __restrict__ props, unsigned int* __restrict__ bpreG) {
    int tid = blockIdx.x * blockDim.x + threadIdx.x;
    if (tid >= NCHUNK * 2 * 32) return;
    int chunk = tid >> 6;
    int t     = (tid >> 5) & 1;
    int L     = tid & 31;
    int colg  = t * 16 + (L & 15);
    int kbase = chunk * 32 + 16 * (L >> 4);
    v16h vals;
#pragma unroll
    for (int j = 0; j < 8; ++j) {
#pragma unroll
        for (int h = 0; h < 2; ++h) {
            int k = kbase + 2 * j + h;
            float v;
            if (colg < C_CH)       v = props[k * 28 + 11 + colg];
            else if (colg == C_CH) v = 1.0f;     // ones column -> wsum
            else                   v = 0.0f;
            vals[2 * j + h] = (_Float16)v;
        }
    }
    *(v16h*)(bpreG + (size_t)tid * 8) = vals;
}

// ---- Main kernel: splat via WMMA + normalization + MLP ----
__global__ __launch_bounds__(256) void splat_main(
    const float* __restrict__ vox,
    const float* __restrict__ geomG,
    const unsigned int* __restrict__ bpreG,
    const float* __restrict__ W1, const float* __restrict__ b1,
    const float* __restrict__ W2, const float* __restrict__ b2,
    float* __restrict__ out)
{
    extern __shared__ char smem[];
    float4*       geomS = (float4*)(smem + GEOM_OFF);
    unsigned int* bpreS = (unsigned int*)(smem + BPRE_OFF);
    float*        scrS  = (float*)(smem + SCR_OFF);
    float*        wS    = (float*)(smem + WTS_OFF);

    int tid = threadIdx.x;

    // ---- Stage L2-resident tables into LDS (192 KB of the 320 KB pool) ----
#if defined(USE_ASYNC_LDS)
    {
        const float4* gg = (const float4*)geomG;
        for (int i = tid; i < NGAUSS * 2; i += 256)
            __builtin_amdgcn_global_load_async_to_lds_b128(
                ASYNC_GPTR(gg + i), ASYNC_LPTR(geomS + i), 0, 0);
        const uint4* bg = (const uint4*)bpreG;
        uint4* bs = (uint4*)bpreS;
        for (int i = tid; i < NCHUNK * 2 * 32 * 2; i += 256)
            __builtin_amdgcn_global_load_async_to_lds_b128(
                ASYNC_GPTR(bg + i), ASYNC_LPTR(bs + i), 0, 0);
    }
#else
    {
        const float4* gg = (const float4*)geomG;
        for (int i = tid; i < NGAUSS * 2; i += 256) geomS[i] = gg[i];
        const uint4* bg = (const uint4*)bpreG;
        uint4* bs = (uint4*)bpreS;
        for (int i = tid; i < NCHUNK * 2 * 32 * 2; i += 256) bs[i] = bg[i];
    }
#endif

    // MLP weights, padded rows (W1: 34x20, W2: 17x36) so ds loads merge to b128.
    for (int i = tid; i < 1360; i += 256) {
        float v = 0.0f;
        if (i < 680)                   { int r = i / 20, c = i % 20; if (c < 17) v = W1[r * 17 + c]; }
        else if (i < 714)              v = b1[i - 680];
        else if (i >= 720 && i < 1332) { int q = i - 720; int r = q / 36, c = q % 36; if (c < 34) v = W2[r * 34 + c]; }
        else if (i >= 1332 && i < 1349) v = b2[i - 1332];
        wS[i] = v;
    }

#if defined(USE_ASYNC_LDS)
#if __has_builtin(__builtin_amdgcn_s_wait_asynccnt)
    __builtin_amdgcn_s_wait_asynccnt(0);
#else
    asm volatile("s_wait_asynccnt 0" ::: "memory");
#endif
#endif
    __syncthreads();

    int lane  = tid & 31;
    int wave  = tid >> 5;
    int m     = lane & 15;       // row within a 16-row M-tile
    int grp   = lane >> 4;       // lane half -> K group / M-row group
    int grp8  = grp * 8;
    int vtile = blockIdx.x * WG_VOX + wave * 32;   // 32 voxels per wave

    int via = vtile + m, vib = vtile + 16 + m;
    float vx0 = vox[via * 3 + 0], vy0 = vox[via * 3 + 1], vz0 = vox[via * 3 + 2];
    float vx1 = vox[vib * 3 + 0], vy1 = vox[vib * 3 + 1], vz1 = vox[vib * 3 + 2];

    v8f c00 = {}, c01 = {};      // voxels 0..15 : ch0-15 | ch16,wsum
    v8f c10 = {}, c11 = {};      // voxels 16..31: ch0-15 | ch16,wsum

#pragma unroll 2
    for (int chunk = 0; chunk < NCHUNK; ++chunk) {
        int kBase = chunk * 32;
        // Pre-swizzled B tiles (shared by both M-tiles): 2x ds_load_b128 each.
        v16h b0  = *(const v16h*)(bpreS + (unsigned)(chunk * 64 + lane) * 8);
        v16h b1v = *(const v16h*)(bpreS + (unsigned)(chunk * 64 + 32 + lane) * 8);

        // Build two A tiles (w matrices, 16x32 f16) in contiguous-K order (same as B, not ISA table order):
        // lanes 0-15: K=0..15; lanes 16-31: K=16..31 (matches B).
        v16h a0, a1;
#pragma unroll
        for (int j = 0; j < 8; ++j) {
#pragma unroll
            for (int h = 0; h < 2; ++h) {
                int kl = 2 * grp8 + 2 * j + h;
                int g  = kBase + kl;
                float4 g0 = geomS[2 * g + 0];     // mean.xyz, inv_s'.x (broadcast)
                float4 g1 = geomS[2 * g + 1];     // inv_s'.yz, log2(op)

                float dx0 = vx0 - g0.x, dy0 = vy0 - g0.y, dz0 = vz0 - g0.z;
                float d2x0 = dx0 * dx0, d2y0 = dy0 * dy0, d2z0 = dz0 * dz0;
                float sq0 = d2x0 + d2y0 + d2z0;
                float e0  = fmaf(d2x0, g0.w, fmaf(d2y0, g1.x, fmaf(d2z0, g1.y, g1.z)));
                float w0  = fast_exp2(e0);
                w0 = (sq0 < 9.0f) ? w0 : 0.0f;
                a0[2 * j + h] = (_Float16)w0;

                float dx1 = vx1 - g0.x, dy1 = vy1 - g0.y, dz1 = vz1 - g0.z;
                float d2x1 = dx1 * dx1, d2y1 = dy1 * dy1, d2z1 = dz1 * dz1;
                float sq1 = d2x1 + d2y1 + d2z1;
                float e1  = fmaf(d2x1, g0.w, fmaf(d2y1, g1.x, fmaf(d2z1, g1.y, g1.z)));
                float w1_ = fast_exp2(e1);
                w1_ = (sq1 < 9.0f) ? w1_ : 0.0f;
                a1[2 * j + h] = (_Float16)w1_;
            }
        }
        c00 = __builtin_amdgcn_wmma_f32_16x16x32_f16(false, a0, false, b0,  (short)0, c00, false, false);
        c01 = __builtin_amdgcn_wmma_f32_16x16x32_f16(false, a0, false, b1v, (short)0, c01, false, false);
        c10 = __builtin_amdgcn_wmma_f32_16x16x32_f16(false, a1, false, b0,  (short)0, c10, false, false);
        c11 = __builtin_amdgcn_wmma_f32_16x16x32_f16(false, a1, false, b1v, (short)0, c11, false, false);
    }

    // Transpose accumulators through per-wave LDS scratch (32 rows x 18).
    // C layout: VGPR r, lanes 0-15 -> M=r, lanes 16-31 -> M=8+r; N = lane&15.
    float* myscr = scrS + wave * (32 * 18);
#pragma unroll
    for (int r = 0; r < 8; ++r) {
        myscr[(grp8 + r) * 18 + m]        = c00[r];
        myscr[(16 + grp8 + r) * 18 + m]   = c10[r];
    }
    if (m < 2) {
#pragma unroll
        for (int r = 0; r < 8; ++r) {
            myscr[(grp8 + r) * 18 + 16 + m]      = c01[r];
            myscr[(16 + grp8 + r) * 18 + 16 + m] = c11[r];
        }
    }
    __syncthreads();

    // Normalize + MLP (17 -> relu(34) -> 17); one voxel per lane (all 32 lanes).
    {
        const float* w1S = wS;            // 34 x 20
        const float* b1S = wS + 680;      // 34
        const float* w2S = wS + 720;      // 17 x 36
        const float* b2S = wS + 1332;     // 17

        const float* row = myscr + lane * 18;
        float inv = 1.0f / fmaxf(row[17], 1e-6f);
        float occ[C_CH];
#pragma unroll
        for (int c = 0; c < C_CH; ++c) occ[c] = row[c] * inv;

        float hbuf[2 * C_CH];
        for (int i = 0; i < 2 * C_CH; ++i) {
            float acc = b1S[i];
            const float* r0 = w1S + i * 20;
#pragma unroll
            for (int c = 0; c < C_CH; ++c) acc = fmaf(occ[c], r0[c], acc);
            hbuf[i] = fmaxf(acc, 0.0f);
        }
        float* o = out + (size_t)(vtile + lane) * C_CH;
        for (int c = 0; c < C_CH; ++c) {
            float acc = b2S[c];
            const float* r2 = w2S + c * 36;
#pragma unroll
            for (int i = 0; i < 2 * C_CH; ++i) acc = fmaf(hbuf[i], r2[i], acc);
            o[c] = acc;
        }
    }
}

// ---------------------------------------------------------------------------
extern "C" void kernel_launch(void* const* d_in, const int* in_sizes, int n_in,
                              void* d_out, int out_size, void* d_ws, size_t ws_size,
                              hipStream_t stream) {
    const float* props = (const float*)d_in[0];   // (1,2048,28)
    const float* vox   = (const float*)d_in[1];   // (640000,3)
    const float* W1    = (const float*)d_in[2];   // (34,17)
    const float* b1    = (const float*)d_in[3];   // (34,)
    const float* W2    = (const float*)d_in[4];   // (17,34)
    const float* b2    = (const float*)d_in[5];   // (17,)
    float* out = (float*)d_out;

    float*        geomG = (float*)d_ws;                            // 64 KB
    unsigned int* bpreG = (unsigned int*)((char*)d_ws + 65536);    // 128 KB

    prep_geom<<<NGAUSS / 256, 256, 0, stream>>>(props, geomG);
    prep_bmat<<<(NCHUNK * 2 * 32) / 256, 256, 0, stream>>>(props, bpreG);

    splat_main<<<NVOX / WG_VOX, 256, SMEM_BYTES, stream>>>(
        vox, geomG, bpreG, W1, b1, W2, b2, out);
}
